// HomogeneousGraphConvolution_38860864094249
// MI455X (gfx1250) — matrix operand order repaired and checked by reading, not compiled, not run on hardware
//
#include <hip/hip_runtime.h>
#include <math.h>

typedef __attribute__((ext_vector_type(16))) _Float16 v16h;
typedef __attribute__((ext_vector_type(2)))  _Float16 v2h;
typedef __attribute__((ext_vector_type(8)))  float    v8f;

#define DFEAT 128
#define LN_EPS 1e-5f

// ---------------------------------------------------------------------------
// Kernel 0: zero the workspace (sums [N*128] + cnt [N]) — grid-stride.
// ---------------------------------------------------------------------------
__global__ void zero_ws_kernel(float* __restrict__ ws, int n) {
  int i = blockIdx.x * blockDim.x + threadIdx.x;
  int stride = gridDim.x * blockDim.x;
  for (; i < n; i += stride) ws[i] = 0.0f;
}

// ---------------------------------------------------------------------------
// Kernel 1: one-time f32 -> f16 weight conversion (done once per launch;
// 64 KB of f16 weights then stay L1/L2-resident for all 6250 GEMM blocks).
// ---------------------------------------------------------------------------
__global__ void convert_w_kernel(const float* __restrict__ Wl,
                                 const float* __restrict__ Wr,
                                 _Float16* __restrict__ Wl16,
                                 _Float16* __restrict__ Wr16) {
  int i = blockIdx.x * blockDim.x + threadIdx.x;
  if (i < DFEAT * DFEAT) {
    Wl16[i] = (_Float16)Wl[i];
    Wr16[i] = (_Float16)Wr[i];
  }
}

// ---------------------------------------------------------------------------
// Kernel 2: edge aggregation. One wave32 per edge; lane l handles 4 floats.
// x[src] is L2-resident (51 MB < 192 MB L2); scatter via f32 L2 atomics.
// ---------------------------------------------------------------------------
__global__ void aggregate_kernel(const float* __restrict__ x,
                                 const int*   __restrict__ ei,
                                 float* __restrict__ sums,
                                 float* __restrict__ cnt,
                                 int E) {
  long long gid = (long long)blockIdx.x * blockDim.x + threadIdx.x;
  int e    = (int)(gid >> 5);
  int lane = (int)(gid & 31);
  if (e >= E) return;
  int src = ei[e];
  int dst = ei[E + e];
  const float4 v = *(const float4*)(x + (size_t)src * DFEAT + lane * 4);
  float* srow = sums + (size_t)dst * DFEAT + lane * 4;
  atomicAdd(srow + 0, v.x);
  atomicAdd(srow + 1, v.y);
  atomicAdd(srow + 2, v.z);
  atomicAdd(srow + 3, v.w);
  if (lane == 0) atomicAdd(cnt + dst, 1.0f);
}

// ---------------------------------------------------------------------------
// Kernel 3: fused  h = mean@W_l^T + b_l + x@W_r^T ; LayerNorm ; exact GELU.
// One block = 16 output rows, 8 waves = 8 column tiles of 16.
// A tile (mean pre-scaled by 1/deg, and x) is staged ONCE per block in LDS
// as f16; each wave then builds WMMA fragments from LDS (broadcast-friendly,
// contiguous 16B runs -> ds_load_b128). B fragments come from the
// pre-converted f16 weights (contiguous 16B runs -> global_load_b128).
// ---------------------------------------------------------------------------
__global__ void __launch_bounds__(256)
sage_fused_kernel(const float* __restrict__ x,
                  const float* __restrict__ sums,
                  const float* __restrict__ cnt,
                  const _Float16* __restrict__ Wl16,
                  const float* __restrict__ bl,
                  const _Float16* __restrict__ Wr16,
                  const float* __restrict__ gamma,
                  const float* __restrict__ beta,
                  float* __restrict__ out,
                  int N) {
  __shared__ _Float16 aM[16 * DFEAT];   // 4 KB: mean rows, f16, pre-scaled
  __shared__ _Float16 aX[16 * DFEAT];   // 4 KB: x rows, f16
  __shared__ float    hlds[16 * DFEAT]; // 8 KB: f32 h tile for LN epilogue

  const int tid  = threadIdx.x;
  const int lane = tid & 31;
  const int wave = tid >> 5;        // 0..7 -> column tile
  const int half = lane >> 4;       // 0/1: which 16-lane half
  const int l16  = lane & 15;

  // ---- Stage A tile in LDS (16 rows x 128, two matrices), f16 ----
  {
    const int srow_i = tid >> 4;          // 0..15: tile row
    const int sseg   = (tid & 15) * 8;    // 8 contiguous cols per thread
    int gr = blockIdx.x * 16 + srow_i;
    if (gr >= N) gr = N - 1;              // clamp: keep EXEC full everywhere
    const float invc = 1.0f / fmaxf(cnt[gr], 1.0f);
    const float* __restrict__ sp = sums + (size_t)gr * DFEAT + sseg;
    const float* __restrict__ xp = x    + (size_t)gr * DFEAT + sseg;
#pragma unroll
    for (int i = 0; i < 8; ++i) {
      aM[srow_i * DFEAT + sseg + i] = (_Float16)(sp[i] * invc);
      aX[srow_i * DFEAT + sseg + i] = (_Float16)xp[i];
    }
  }
  __syncthreads();

  const int col = wave * 16 + l16;              // global output column j
  const _Float16* __restrict__ amrow = aM + l16 * DFEAT;
  const _Float16* __restrict__ axrow = aX + l16 * DFEAT;
  const _Float16* __restrict__ wlr   = Wl16 + (size_t)col * DFEAT; // W_l[j][k]
  const _Float16* __restrict__ wrr   = Wr16 + (size_t)col * DFEAT; // W_r[j][k]

  v8f acc = {};

#pragma unroll
  for (int kc = 0; kc < 4; ++kc) {
    v16h am, ax, bwl, bwr;
    // A fragment (16x32 f16): VGPR v holds K = {2v | 16+2(v-4)} + 8*half
#pragma unroll
    for (int v = 0; v < 8; ++v) {
      const int k = kc * 32 + ((v < 4) ? 2 * v : 16 + 2 * (v - 4)) + 8 * half;
      const v2h pm = *(const v2h*)(amrow + k);
      const v2h px = *(const v2h*)(axrow + k);
      am[2 * v]     = pm[0];
      am[2 * v + 1] = pm[1];
      ax[2 * v]     = px[0];
      ax[2 * v + 1] = px[1];
    }
    // B fragment (32x16 f16): lane col j fixed, VGPR i holds K = {2i | 16+2(i-4)} + 8*half
#pragma unroll
    for (int i = 0; i < 8; ++i) {
      const int k = kc * 32 + ((i < 4) ? 2 * i : 16 + 2 * (i - 4)) + 8 * half;
      const v2h pl = *(const v2h*)(wlr + k);
      const v2h pr = *(const v2h*)(wrr + k);
      bwl[2 * i]     = pl[0];
      bwl[2 * i + 1] = pl[1];
      bwr[2 * i]     = pr[0];
      bwr[2 * i + 1] = pr[1];
    }
    acc = __builtin_amdgcn_wmma_f32_16x16x32_f16(false, am, false, bwl,
                                                 (short)0, acc, false, false);
    acc = __builtin_amdgcn_wmma_f32_16x16x32_f16(false, ax, false, bwr,
                                                 (short)0, acc, false, false);
  }

  // C/D layout: VGPR r -> row M = r + 8*half, col N = lane&15. Add bias, stage.
  const float bias = bl[col];
#pragma unroll
  for (int r = 0; r < 8; ++r) {
    hlds[(r + 8 * half) * DFEAT + col] = acc[r] + bias;
  }
  __syncthreads();

  // LayerNorm + exact GELU: 16 threads per row; shfl tree inside 16-lane half.
  const int row = tid >> 4;   // 0..15
  const int sub = tid & 15;
  float s = 0.0f, ss = 0.0f;
#pragma unroll
  for (int i = 0; i < 8; ++i) {
    const float h = hlds[row * DFEAT + sub + 16 * i];
    s += h;
    ss += h * h;
  }
#pragma unroll
  for (int m = 8; m >= 1; m >>= 1) {
    s  += __shfl_xor(s, m, 32);
    ss += __shfl_xor(ss, m, 32);
  }
  const float mu   = s * (1.0f / DFEAT);
  const float var  = ss * (1.0f / DFEAT) - mu * mu;
  const float rstd = rsqrtf(var + LN_EPS);

  const int out_row = blockIdx.x * 16 + row;
  if (out_row < N) {
    float* __restrict__ orow = out + (size_t)out_row * DFEAT;
#pragma unroll
    for (int i = 0; i < 8; ++i) {
      const int j = sub + 16 * i;
      const float hn = (hlds[row * DFEAT + j] - mu) * rstd * gamma[j] + beta[j];
      orow[j] = 0.5f * hn * (1.0f + erff(hn * 0.70710678118654752f));
    }
  }
}

// ---------------------------------------------------------------------------
extern "C" void kernel_launch(void* const* d_in, const int* in_sizes, int n_in,
                              void* d_out, int out_size, void* d_ws, size_t ws_size,
                              hipStream_t stream) {
  const float* x     = (const float*)d_in[0];
  const int*   ei    = (const int*)  d_in[1];
  const float* Wl    = (const float*)d_in[2];
  const float* blv   = (const float*)d_in[3];
  const float* Wr    = (const float*)d_in[4];
  const float* gamma = (const float*)d_in[5];
  const float* beta  = (const float*)d_in[6];
  float*       out   = (float*)d_out;

  const int N = in_sizes[0] / DFEAT;
  const int E = in_sizes[1] / 2;

  // Workspace layout: sums [N*128 f32] | cnt [N f32] | Wl16 | Wr16 (f16)
  float*    sums = (float*)d_ws;
  float*    cnt  = sums + (size_t)N * DFEAT;
  _Float16* Wl16 = (_Float16*)(cnt + N);
  _Float16* Wr16 = Wl16 + DFEAT * DFEAT;

  // 1) zero sums + cnt
  const int nz = N * (DFEAT + 1);
  zero_ws_kernel<<<4096, 256, 0, stream>>>(sums, nz);

  // 2) one-time weight conversion to f16
  convert_w_kernel<<<(DFEAT * DFEAT + 255) / 256, 256, 0, stream>>>(Wl, Wr,
                                                                    Wl16, Wr16);

  // 3) edge aggregation (one wave32 per edge)
  const long long agg_threads = (long long)E * 32;
  const int agg_blocks = (int)((agg_threads + 255) / 256);
  aggregate_kernel<<<agg_blocks, 256, 0, stream>>>(x, ei, sums, cnt, E);

  // 4) fused WMMA GEMM + LayerNorm + GELU
  const int gemm_blocks = (N + 15) / 16;
  sage_fused_kernel<<<gemm_blocks, 256, 0, stream>>>(x, sums, cnt, Wl16, blv,
                                                     Wr16, gamma, beta, out, N);
}
